// NonLocalBlock_89464168776015
// MI455X (gfx1250) — matrix operand order repaired and checked by reading, not compiled, not run on hardware
//
#include <hip/hip_runtime.h>
#include <math.h>

// NonLocalBlock fused pipeline for MI455X (gfx1250, wave32, WMMA bf16 16x16x32).
// B=4, C=256, Ci=128, N=64*64=4096.

#define DEVINL __device__ __forceinline__

typedef __attribute__((ext_vector_type(16))) __bf16 bf16x16;
typedef __attribute__((ext_vector_type(8)))  __bf16 bf16x8;
typedef __attribute__((ext_vector_type(8)))  float  f32x8;
typedef __attribute__((ext_vector_type(4)))  int    v4i;

namespace {
constexpr int BB = 4;
constexpr int C  = 256;
constexpr int CI = 128;
constexpr int N  = 4096;
}

// Async global->LDS (gfx1250). Guarded: fall back to synchronous LDS staging
// if the toolchain lacks the builtin.
#if __has_builtin(__builtin_amdgcn_global_load_async_to_lds_b128)
#define NLB_ASYNC 1
#endif

DEVINL void cp16_to_lds(const __bf16* src, __bf16* dst) {
#ifdef NLB_ASYNC
  typedef __attribute__((address_space(1))) v4i* gv4i_p;   // global (AS1)
  typedef __attribute__((address_space(3))) v4i* lv4i_p;   // LDS (AS3)
  __builtin_amdgcn_global_load_async_to_lds_b128((gv4i_p)src, (lv4i_p)dst, 0, 0);
#else
  *(bf16x8*)dst = *(const bf16x8*)src;
#endif
}

DEVINL void wait_async_all() {
#ifdef NLB_ASYNC
#if __has_builtin(__builtin_amdgcn_s_wait_asynccnt)
  __builtin_amdgcn_s_wait_asynccnt(0);
#else
  asm volatile("s_wait_asynccnt 0" ::: "memory");
#endif
#endif
}

DEVINL f32x8 zero8() {
  f32x8 z;
#pragma unroll
  for (int i = 0; i < 8; ++i) z[i] = 0.0f;
  return z;
}

// A-fragment, 16x32 bf16, memory row-major [16 rows][ld], rows contiguous in K.
// Contiguous-K order used here (same as load_b32x16): lanes 0-15 hold row=lane, K=0..15; lanes 16-31 hold
// row=lane-16, K=16..31.
DEVINL bf16x16 load_a16x32(const __bf16* tile, int ld) {
  const int lane = threadIdx.x & 31;
  const int row  = lane & 15;
  const int half = lane >> 4;
  const __bf16* p = tile + (size_t)row * ld + half * 16;
  bf16x8 lo = *(const bf16x8*)(p);
  bf16x8 hi = *(const bf16x8*)(p + 8);
  bf16x16 r;
#pragma unroll
  for (int i = 0; i < 8; ++i) { r[i] = lo[i]; r[8 + i] = hi[i]; }
  return r;
}

// B-fragment, 32x16 bf16, memory holds B^T row-major: [16 cols][ld] contiguous K.
// Contiguous-K order used here (same as load_a16x32): lanes 0-15: col=lane, K=0..15; lanes 16-31: col=lane-16, K=16..31.
DEVINL bf16x16 load_b32x16(const __bf16* tileT, int ld) {
  const int lane = threadIdx.x & 31;
  const int col  = lane & 15;
  const int half = lane >> 4;
  const __bf16* p = tileT + (size_t)col * ld + half * 16;
  bf16x8 a = *(const bf16x8*)(p);
  bf16x8 b = *(const bf16x8*)(p + 8);
  bf16x16 r;
#pragma unroll
  for (int i = 0; i < 8; ++i) { r[i] = a[i]; r[8 + i] = b[i]; }
  return r;
}

DEVINL f32x8 wmma_bf16(bf16x16 a, bf16x16 b, f32x8 c) {
  return __builtin_amdgcn_wmma_f32_16x16x32_bf16(false, a, false, b, (short)0, c,
                                                 false, false);
}

// Exchange with lane^16 (SWAPX16 group-of-32 swizzle: xor=0x10, and=0x1f).
DEVINL float swz_xor16(float x) {
  return __int_as_float(__builtin_amdgcn_ds_swizzle(__float_as_int(x), 0x401f));
}

// ---------------------------------------------------------------------------
// Kernel 0: convert weights to bf16 (theta_w/phi_w/g_w [Ci,C], out_w [C,Ci]).
__global__ __launch_bounds__(256) void nlb_prep_w(
    const float* __restrict__ tw, const float* __restrict__ pw,
    const float* __restrict__ gw, const float* __restrict__ ow,
    __bf16* __restrict__ wq, __bf16* __restrict__ wk,
    __bf16* __restrict__ wv, __bf16* __restrict__ wo) {
  int i = blockIdx.x * 256 + threadIdx.x;
  if (i < CI * C) {
    wq[i] = (__bf16)tw[i];
    wk[i] = (__bf16)pw[i];
    wv[i] = (__bf16)gw[i];
    wo[i] = (__bf16)ow[i];  // C*Ci == Ci*C elements
  }
}

// ---------------------------------------------------------------------------
// Kernel 1: x [B,C,N] f32 -> xT [B,N,C] bf16 (LDS tiled transpose).
__global__ __launch_bounds__(256) void nlb_transpose_x(
    const float* __restrict__ x, __bf16* __restrict__ xT) {
  __shared__ float t[32][33];
  const int b  = blockIdx.z;
  const int c0 = blockIdx.y * 32;
  const int n0 = blockIdx.x * 32;
  const int tx = threadIdx.x, ty = threadIdx.y;  // (32, 8)
  const float* xp = x + (size_t)b * C * N;
#pragma unroll
  for (int i = 0; i < 32; i += 8)
    t[ty + i][tx] = xp[(size_t)(c0 + ty + i) * N + n0 + tx];
  __syncthreads();
  __bf16* o = xT + (size_t)b * N * C;
#pragma unroll
  for (int i = 0; i < 32; i += 8)
    o[(size_t)(n0 + ty + i) * C + c0 + tx] = (__bf16)t[tx][ty + i];
}

// ---------------------------------------------------------------------------
// Kernel 2: fused theta/phi/g projections via WMMA.
//   thetaT/phiT: [B,N,Ci] bf16 (row n contiguous ci)   g: [B,Ci,N] bf16
__global__ __launch_bounds__(256) void nlb_proj(
    const __bf16* __restrict__ xT,
    const __bf16* __restrict__ wq, const __bf16* __restrict__ wk,
    const __bf16* __restrict__ wv,
    const float* __restrict__ bq, const float* __restrict__ bk,
    const float* __restrict__ bv,
    __bf16* __restrict__ thetaT, __bf16* __restrict__ phiT,
    __bf16* __restrict__ gmat) {
  const int wave = threadIdx.x >> 5;
  int t = blockIdx.x * 8 + wave;          // B*3*(N/16)*(Ci/16) = 24576 tiles
  const int ciT = t & 7;   t >>= 3;
  const int nT  = t & 255; t >>= 8;
  const int proj = t % 3;
  const int b    = t / 3;

  const __bf16* w   = (proj == 0) ? wq : ((proj == 1) ? wk : wv);
  const float* bias = (proj == 0) ? bq : ((proj == 1) ? bk : bv);
  const __bf16* xa  = xT + ((size_t)b * N + nT * 16) * C;

  f32x8 acc = zero8();
#pragma unroll
  for (int k = 0; k < C; k += 32) {
    bf16x16 a  = load_a16x32(xa + k, C);                     // x^T tile [16n x 32c]
    bf16x16 bb = load_b32x16(w + (size_t)(ciT * 16) * C + k, C);  // W rows = B^T
    acc = wmma_bf16(a, bb, acc);
  }
  const int lane = threadIdx.x & 31;
  const int col  = lane & 15;          // ci within tile
  const int rb   = (lane >> 4) * 8;    // n-row base within tile
  const int ci   = ciT * 16 + col;
  const float bv_ = bias[ci];
  if (proj < 2) {
    __bf16* out = ((proj == 0) ? thetaT : phiT) + (size_t)b * N * CI;
#pragma unroll
    for (int v = 0; v < 8; ++v)
      out[(size_t)(nT * 16 + rb + v) * CI + ci] = (__bf16)(acc[v] + bv_);
  } else {
    __bf16* out = gmat + (size_t)b * CI * N;
#pragma unroll
    for (int v = 0; v < 8; ++v)
      out[(size_t)ci * N + nT * 16 + rb + v] = (__bf16)(acc[v] + bv_);
  }
}

// ---------------------------------------------------------------------------
// Kernel 3: fused flash attention with double-buffered async LDS staging of
// the K/V tiles shared by all 4 waves of the block. Computes S^T so each lane
// owns one query row n; online softmax is per-lane + one xor16 swizzle.
// O^T[ci,n] accumulated via g-tiles x P^T; oT written [B,N,Ci] bf16.
__global__ __launch_bounds__(128) void nlb_attn(
    const __bf16* __restrict__ thetaT, const __bf16* __restrict__ phiT,
    const __bf16* __restrict__ gmat, __bf16* __restrict__ oT) {
  // double-buffered K/V tiles: phi [32 m][128 ci], g [128 ci][32 m]
  __shared__ __bf16 sphi[2][32 * CI];
  __shared__ __bf16 sg[2][CI * 32];

  const int tid   = threadIdx.x;               // 0..127
  const int wave  = tid >> 5;
  const int lane  = tid & 31;
  const int col   = lane & 15;                 // query row n within tile
  const int half  = lane >> 4;
  const int b     = blockIdx.y;
  const int nBase = (blockIdx.x * 4 + wave) * 16;

  const __bf16* th = thetaT + ((size_t)b * N + nBase) * CI;
  const __bf16* ph = phiT + (size_t)b * N * CI;
  const __bf16* gm = gmat + (size_t)b * CI * N;

  // Stage one 32-key tile (16 KB) cooperatively: 8 x b128 per thread.
  auto stage = [&](int buf, int m0) {
    // phi rows m0..m0+31 are one contiguous 8 KB block of phiT
    const __bf16* psrc = ph + (size_t)m0 * CI;
#pragma unroll
    for (int j = 0; j < 4; ++j) {
      const int e = (tid + j * 128) * 8;       // 16-byte chunk
      cp16_to_lds(psrc + e, &sphi[buf][e]);
    }
    // g: row ci = tid (128 rows), 4 chunks of 8 elements each
    const __bf16* gsrc = gm + (size_t)tid * N + m0;
#pragma unroll
    for (int j = 0; j < 4; ++j)
      cp16_to_lds(gsrc + j * 8, &sg[buf][tid * 32 + j * 8]);
  };

  // Q (theta rows) as B-fragments over Ci (4 K-steps of 32), held in VGPRs.
  bf16x16 q[4];
#pragma unroll
  for (int k = 0; k < 4; ++k) q[k] = load_b32x16(th + k * 32, CI);

  f32x8 acc[8];                                // O^T: 8 ci-tiles of 16
#pragma unroll
  for (int tIdx = 0; tIdx < 8; ++tIdx) acc[tIdx] = zero8();
  float m_run = -1.0e30f;
  float l_run = 0.0f;
  const bool low = lane < 16;

  stage(0, 0);
  wait_async_all();
  __syncthreads();

  for (int m0 = 0; m0 < N; m0 += 32) {
    const int cur = (m0 >> 5) & 1;
    if (m0 + 32 < N) stage(cur ^ 1, m0 + 32);  // async prefetch next tile

    // ---- S^T tiles (rows m, cols n) from LDS ----
    f32x8 st0 = zero8(), st1 = zero8();
#pragma unroll
    for (int k = 0; k < 4; ++k) {
      bf16x16 a0 = load_a16x32(&sphi[cur][k * 32], CI);
      bf16x16 a1 = load_a16x32(&sphi[cur][16 * CI + k * 32], CI);
      st0 = wmma_bf16(a0, q[k], st0);
      st1 = wmma_bf16(a1, q[k], st1);
    }
    // ---- online softmax over the 32 m values of this lane's column n ----
    float mx = -1.0e30f;
#pragma unroll
    for (int v = 0; v < 8; ++v) {
      mx = fmaxf(mx, st0[v]);
      mx = fmaxf(mx, st1[v]);
    }
    mx = fmaxf(mx, swz_xor16(mx));             // combine the two lane halves
    const float m_new = fmaxf(m_run, mx);
    const float corr  = __expf(m_run - m_new);
    f32x8 p0, p1;
    float psum = 0.0f;
#pragma unroll
    for (int v = 0; v < 8; ++v) {
      p0[v] = __expf(st0[v] - m_new);
      p1[v] = __expf(st1[v] - m_new);
      psum += p0[v] + p1[v];
    }
    psum += swz_xor16(psum);
    l_run = l_run * corr + psum;
    m_run = m_new;
#pragma unroll
    for (int tIdx = 0; tIdx < 8; ++tIdx) acc[tIdx] = acc[tIdx] * corr;

    // ---- repack P^T into B-fragment [K=32 m x 16 n] via one xor16 swap ----
    // low lanes need m 0..15 (own m=v, partner m=8+v); high lanes need
    // m 16..31 (partner m=16+v, own m=24+v).
    bf16x16 pf;
#pragma unroll
    for (int v = 0; v < 8; ++v) {
      float o0 = swz_xor16(p0[v]);
      float o1 = swz_xor16(p1[v]);
      pf[v]     = (__bf16)(low ? p0[v] : o1);
      pf[8 + v] = (__bf16)(low ? o0 : p1[v]);
    }
    // ---- O^T += g_tile x P^T (g fragments from LDS) ----
#pragma unroll
    for (int tIdx = 0; tIdx < 8; ++tIdx) {
      bf16x16 ga = load_a16x32(&sg[cur][tIdx * 16 * 32], 32);
      acc[tIdx] = wmma_bf16(ga, pf, acc[tIdx]);
    }

    wait_async_all();      // next tile's DMA done (this wave's issues)
    __syncthreads();       // ...and everyone's; also fences buffer reuse
  }

  // normalize and store oT[n, ci] (8 contiguous bf16 per tile per lane)
  const float inv = 1.0f / l_run;
  __bf16* op = oT + ((size_t)b * N + nBase + col) * CI + half * 8;
#pragma unroll
  for (int tIdx = 0; tIdx < 8; ++tIdx) {
    bf16x8 s;
#pragma unroll
    for (int v = 0; v < 8; ++v) s[v] = (__bf16)(acc[tIdx][v] * inv);
    *(bf16x8*)(op + tIdx * 16) = s;
  }
}

// ---------------------------------------------------------------------------
// Kernel 4: y[c,n] = out_w[c,:] . o[:,n] + out_b[c] + x[c,n], LDS-bounced.
__global__ __launch_bounds__(256) void nlb_outproj(
    const __bf16* __restrict__ oT, const __bf16* __restrict__ wo,
    const float* __restrict__ ob, const float* __restrict__ x,
    float* __restrict__ y) {
  __shared__ float tile[8][16][17];
  const int wave = threadIdx.x >> 5;
  const int lane = threadIdx.x & 31;
  int t = blockIdx.x * 8 + wave;        // B*(C/16)*(N/16) = 16384 tiles
  const int nT = t & 255; t >>= 8;
  const int cT = t & 15;  t >>= 4;
  const int b  = t;

  f32x8 acc = zero8();
#pragma unroll
  for (int k = 0; k < CI; k += 32) {
    bf16x16 a  = load_a16x32(wo + (size_t)(cT * 16) * CI + k, CI);  // [16c x 32ci]
    bf16x16 bb = load_b32x16(oT + ((size_t)b * N + nT * 16) * CI + k, CI);
    acc = wmma_bf16(a, bb, acc);
  }
  const int col = lane & 15;           // n within tile
  const int rb  = (lane >> 4) * 8;     // c-row base
#pragma unroll
  for (int v = 0; v < 8; ++v) tile[wave][rb + v][col] = acc[v];
  __syncthreads();

  const int c = cT * 16 + (lane & 15);
  const int h = lane >> 4;
  const size_t base = ((size_t)b * C + c) * N + nT * 16 + h * 8;
  const float bias = ob[c];
#pragma unroll
  for (int j = 0; j < 8; ++j)
    y[base + j] = tile[wave][lane & 15][h * 8 + j] + bias + x[base + j];
}

// ---------------------------------------------------------------------------
extern "C" void kernel_launch(void* const* d_in, const int* in_sizes, int n_in,
                              void* d_out, int out_size, void* d_ws, size_t ws_size,
                              hipStream_t stream) {
  const float* x       = (const float*)d_in[0];
  const float* theta_w = (const float*)d_in[1];
  const float* theta_b = (const float*)d_in[2];
  const float* phi_w   = (const float*)d_in[3];
  const float* phi_b   = (const float*)d_in[4];
  const float* g_w     = (const float*)d_in[5];
  const float* g_b     = (const float*)d_in[6];
  const float* out_w   = (const float*)d_in[7];
  const float* out_b   = (const float*)d_in[8];
  float* y = (float*)d_out;

  char* ws = (char*)d_ws;
  size_t off = 0;
  __bf16* xT = (__bf16*)(ws + off); off += (size_t)BB * N * C * 2;   // 8 MB
  __bf16* wq = (__bf16*)(ws + off); off += (size_t)CI * C * 2;
  __bf16* wk = (__bf16*)(ws + off); off += (size_t)CI * C * 2;
  __bf16* wv = (__bf16*)(ws + off); off += (size_t)CI * C * 2;
  __bf16* wo = (__bf16*)(ws + off); off += (size_t)C * CI * 2;
  __bf16* thetaT = (__bf16*)(ws + off); off += (size_t)BB * N * CI * 2;  // 4 MB
  __bf16* phiT   = (__bf16*)(ws + off); off += (size_t)BB * N * CI * 2;  // 4 MB
  __bf16* gmat   = (__bf16*)(ws + off); off += (size_t)BB * CI * N * 2;  // 4 MB
  __bf16* oT     = (__bf16*)(ws + off); off += (size_t)BB * N * CI * 2;  // 4 MB

  nlb_prep_w<<<(CI * C + 255) / 256, 256, 0, stream>>>(
      theta_w, phi_w, g_w, out_w, wq, wk, wv, wo);
  nlb_transpose_x<<<dim3(N / 32, C / 32, BB), dim3(32, 8), 0, stream>>>(x, xT);
  nlb_proj<<<(BB * 3 * (N / 16) * (CI / 16)) / 8, 256, 0, stream>>>(
      xT, wq, wk, wv, theta_b, phi_b, g_b, thetaT, phiT, gmat);
  nlb_attn<<<dim3(N / 64, BB), 128, 0, stream>>>(thetaT, phiT, gmat, oT);
  nlb_outproj<<<(BB * (C / 16) * (N / 16)) / 8, 256, 0, stream>>>(
      oT, wo, out_b, x, y);
}
